// SlidingBaseFc_74706661146967
// MI455X (gfx1250) — matrix operand order repaired and checked by reading, not compiled, not run on hardware
//
#include <hip/hip_runtime.h>

typedef __attribute__((ext_vector_type(16))) _Float16 v16h;
typedef __attribute__((ext_vector_type(4)))  _Float16 v4h;
typedef __attribute__((ext_vector_type(2)))  _Float16 v2h;
typedef __attribute__((ext_vector_type(8)))  float    v8f;
typedef __attribute__((ext_vector_type(4)))  float    v4f;

#define BM 64
#define BN 128
#define BK 32
#define LDA_S (BK + 8)    // 40 halves per LDS row (80B)

// ---------------------------------------------------------------------------
// GEMM: Out[1024 x N] = act(A[1024 x Kp](f32, lda%4==0, Kp%32==0, zero-padded)
//                            @ Wsw (pre-swizzled f16 B fragments) + bias)
// Wsw layout: [kb][ntile][lane(32)][j(16)] halves, kb=K/32 blocks, ntile=N/16.
//   lane 0-15  hold column n = ntile*16 + lane, K = kb*32 + j      (j=0..15)
//   lane 16-31 hold column n = ntile*16 + lane-16, K = kb*32+16+j
// ---------------------------------------------------------------------------
__global__ __launch_bounds__(256) void gemm_bias_act_wmma(
    const float* __restrict__ A, int lda,
    const _Float16* __restrict__ Wsw, int NT,
    const float* __restrict__ bias,
    float* __restrict__ Out, int ldo,
    int N, int Kp, int relu)
{
  __shared__ _Float16 As[BM * LDA_S];

  const int tid  = threadIdx.x;
  const int lane = tid & 31;
  const int wave = tid >> 5;            // 0..7
  const int wm   = wave >> 2;           // 0..1 (M direction)
  const int wn   = wave & 3;            // 0..3 (N direction)
  const int m0   = blockIdx.y * BM;
  const int n0   = blockIdx.x * BN;

  // clamp padded n-tiles into range (results discarded by store guard)
  const int tg0 = (n0 >> 4) + wn * 2;
  const int NTm1 = NT - 1;
  const int ntc0 = (tg0     < NTm1) ? tg0     : NTm1;
  const int ntc1 = (tg0 + 1 < NTm1) ? tg0 + 1 : NTm1;

  v8f acc[2][2] = {};

  const int mr = lane & 15;
  const int kA = (lane < 16) ? 0 : 16;  // A-frag: VGPR0..7 = K{kA..kA+15} pairs

  int kb = 0;
  for (int k0 = 0; k0 < Kp; k0 += BK, ++kb) {
    // ---- stage A tile (64x32 f32 -> f16), unconditional vector path ----
#pragma unroll
    for (int i = 0; i < 2; ++i) {
      int g = i * 256 + tid;            // 0..511
      int r = g >> 3;                   // 0..63
      int c = (g & 7) << 2;             // 0..28
      v4f fv = *(const v4f*)(A + (size_t)(m0 + r) * lda + (k0 + c));
      v4h hv;
      hv[0] = (_Float16)fv[0]; hv[1] = (_Float16)fv[1];
      hv[2] = (_Float16)fv[2]; hv[3] = (_Float16)fv[3];
      *(v4h*)&As[r * LDA_S + c] = hv;
    }
    __syncthreads();

    // ---- A fragments from LDS (2x ds_load_b128 each) ----
    v16h af[2];
#pragma unroll
    for (int mt = 0; mt < 2; ++mt) {
      const _Float16* ap = &As[(wm * 32 + mt * 16 + mr) * LDA_S];
#pragma unroll
      for (int v = 0; v < 8; ++v) {
        int kk = kA + 2 * v;
        v2h p = *(const v2h*)(ap + kk);
        af[mt][2 * v]     = p[0];
        af[mt][2 * v + 1] = p[1];
      }
    }

    // ---- B fragments straight from pre-swizzled global (L2-resident) ----
    v16h bf[2];
    bf[0] = *(const v16h*)(Wsw + (((size_t)kb * NT + ntc0) * 32 + lane) * 16);
    bf[1] = *(const v16h*)(Wsw + (((size_t)kb * NT + ntc1) * 32 + lane) * 16);

#pragma unroll
    for (int mt = 0; mt < 2; ++mt) {
      acc[mt][0] = __builtin_amdgcn_wmma_f32_16x16x32_f16(
          false, af[mt], false, bf[0], (short)0, acc[mt][0], false, false);
      acc[mt][1] = __builtin_amdgcn_wmma_f32_16x16x32_f16(
          false, af[mt], false, bf[1], (short)0, acc[mt][1], false, false);
    }
    __syncthreads();
  }

  // ---- epilogue: C/D layout: VGPR r, lanes<16 -> M=r, lanes>=16 -> M=r+8; N=lane&15 ----
  const int mloc = (lane >> 4) << 3;
  const int nloc = lane & 15;
#pragma unroll
  for (int mt = 0; mt < 2; ++mt) {
#pragma unroll
    for (int nt = 0; nt < 2; ++nt) {
      int gn = n0 + wn * 32 + nt * 16 + nloc;
      float bv = (gn < N) ? bias[gn] : 0.0f;
#pragma unroll
      for (int r = 0; r < 8; ++r) {
        int gm = m0 + wm * 32 + mt * 16 + r + mloc;
        if (gn < N) {
          float v = acc[mt][nt][r] + bv;
          if (relu) v = fmaxf(v, 0.0f);
          Out[(size_t)gm * ldo + gn] = v;
        }
      }
    }
  }
}

// Swizzle W (K x N, f32 row-major) into WMMA B-fragment order (f16, zero-padded).
__global__ void swizzle_w(const float* __restrict__ W, _Float16* __restrict__ Wsw,
                          int K, int N, int NT, int total) {
  int i = blockIdx.x * blockDim.x + threadIdx.x;
  if (i >= total) return;
  int j    = i & 15;
  int lane = (i >> 4) & 31;
  int tile = i >> 9;                   // kb * NT + ntile
  int ntile = tile % NT;
  int kb    = tile / NT;
  int k = kb * 32 + ((lane < 16) ? 0 : 16) + j;
  int n = ntile * 16 + (lane & 15);
  Wsw[i] = (k < K && n < N) ? (_Float16)W[(size_t)k * N + n] : (_Float16)0.0f;
}

// source (1024 x 3060) -> padded (1024 x 3072), zero tail
__global__ void pad_source(const float* __restrict__ src, float* __restrict__ dst) {
  int i = blockIdx.x * blockDim.x + threadIdx.x;
  if (i >= 1024 * 3072) return;
  int b = i >> 11;  // /3072? 3072 = 2^10*3 -> not power of two; compute properly
  b = i / 3072;
  int j = i - b * 3072;
  dst[i] = (j < 3060) ? src[(size_t)b * 3060 + j] : 0.0f;
}

// feat[b, 0:36]=src[b,t-1,:]; [36:72]=src[b,t,:]; [72:141]=result[b,t-1,:]; [141:160]=0
__global__ void concat_feat(const float* __restrict__ src, const float* __restrict__ res,
                            float* __restrict__ feat, int t) {
  int i = blockIdx.x * blockDim.x + threadIdx.x;
  if (i >= 1024 * 160) return;
  int b = i / 160, j = i - b * 160;
  float v;
  if (j < 36)       v = src[(size_t)b * 3060 + (size_t)(t - 1) * 36 + j];
  else if (j < 72)  v = src[(size_t)b * 3060 + (size_t)t * 36 + (j - 36)];
  else if (j < 141) v = res[(size_t)b * 5865 + (size_t)(t - 1) * 69 + (j - 72)];
  else              v = 0.0f;
  feat[i] = v;
}

__global__ void copy_init(const float* __restrict__ res, float* __restrict__ init_out) {
  int i = blockIdx.x * blockDim.x + threadIdx.x;
  if (i >= 1024 * 69) return;
  int b = i / 69, c = i - b * 69;
  init_out[i] = res[(size_t)b * 5865 + c];
}

extern "C" void kernel_launch(void* const* d_in, const int* in_sizes, int n_in,
                              void* d_out, int out_size, void* d_ws, size_t ws_size,
                              hipStream_t stream) {
  (void)in_sizes; (void)n_in; (void)out_size; (void)ws_size;
  const float* src  = (const float*)d_in[0];
  const float* wi1  = (const float*)d_in[1];  const float* bi1  = (const float*)d_in[2];
  const float* wi2  = (const float*)d_in[3];  const float* bi2  = (const float*)d_in[4];
  const float* wi25 = (const float*)d_in[5];  const float* bi25 = (const float*)d_in[6];
  const float* wi3  = (const float*)d_in[7];  const float* bi3  = (const float*)d_in[8];
  const float* w1   = (const float*)d_in[9];  const float* b1   = (const float*)d_in[10];
  const float* w12  = (const float*)d_in[11]; const float* b12  = (const float*)d_in[12];
  const float* w2   = (const float*)d_in[13]; const float* b2   = (const float*)d_in[14];
  const float* w22  = (const float*)d_in[15]; const float* b22  = (const float*)d_in[16];
  const float* w3   = (const float*)d_in[17]; const float* b3   = (const float*)d_in[18];
  const float* w4   = (const float*)d_in[19]; const float* b4   = (const float*)d_in[20];
  const float* w4d  = (const float*)d_in[21]; const float* b4d  = (const float*)d_in[22];

  float* result   = (float*)d_out;                       // (1024, 85, 69), row stride 5865
  float* init_out = result + (size_t)1024 * 85 * 69;     // (1024, 69)

  char* ws = (char*)d_ws;
  size_t off = 0;
  auto allocH = [&](size_t n) {
    _Float16* p = (_Float16*)(ws + off);
    off = (off + n * 2 + 255) & ~(size_t)255;
    return p;
  };
  auto allocF = [&](size_t n) {
    float* p = (float*)(ws + off);
    off = (off + n * 4 + 255) & ~(size_t)255;
    return p;
  };

  // swizzled weight sizes: KB*NT*512 halves (KB=ceil(K/32), NT=ceil(N/16))
  struct WDesc { const float* w; int K, N, KB, NT; _Float16* d; };
  WDesc wd[11] = {
    { wi1, 3060, 2048, 96, 128, nullptr },
    { wi2, 2048, 1024, 64,  64, nullptr },
    { wi25,1024,  512, 32,  32, nullptr },
    { wi3,  512,   69, 16,   5, nullptr },
    { w1,   141, 2048,  5, 128, nullptr },
    { w12, 2048, 2048, 64, 128, nullptr },
    { w2,  2048, 1024, 64,  64, nullptr },
    { w22, 1024, 1024, 32,  64, nullptr },
    { w3,  1024,  512, 32,  32, nullptr },
    { w4,   512,  512, 16,  32, nullptr },
    { w4d,  512,   69, 16,   5, nullptr },
  };
  for (int i = 0; i < 11; ++i)
    wd[i].d = allocH((size_t)wd[i].KB * wd[i].NT * 512);

  float* srcP = allocF((size_t)1024 * 3072);
  float* bufA = allocF((size_t)1024 * 2048);
  float* bufB = allocF((size_t)1024 * 2048);
  float* feat = allocF((size_t)1024 * 160);

  for (int i = 0; i < 11; ++i) {
    int total = wd[i].KB * wd[i].NT * 512;
    swizzle_w<<<(total + 255) / 256, 256, 0, stream>>>(
        wd[i].w, wd[i].d, wd[i].K, wd[i].N, wd[i].NT, total);
  }
  pad_source<<<(1024 * 3072 + 255) / 256, 256, 0, stream>>>(src, srcP);

  auto gemm = [&](const float* Aa, int lda, int widx, const float* bb,
                  float* Oo, int ldo, int Nn, int relu) {
    int Kp = wd[widx].KB * 32;
    dim3 g((Nn + BN - 1) / BN, 1024 / BM);
    gemm_bias_act_wmma<<<g, dim3(256), 0, stream>>>(
        Aa, lda, wd[widx].d, wd[widx].NT, bb, Oo, ldo, Nn, Kp, relu);
  };

  // ---- init-pos MLP ----
  gemm(srcP, 3072, 0, bi1,  bufA,   2048, 2048, 1);
  gemm(bufA, 2048, 1, bi2,  bufB,   1024, 1024, 1);
  gemm(bufB, 1024, 2, bi25, bufA,    512,  512, 0);
  gemm(bufA,  512, 3, bi3,  result, 5865,   69, 0);  // result[:,0,:]
  copy_init<<<(1024 * 69 + 255) / 256, 256, 0, stream>>>(result, init_out);

  // ---- sequential scan over t = 1..84 (stream order carries the dependency) ----
  for (int t = 1; t < 85; ++t) {
    concat_feat<<<(1024 * 160 + 255) / 256, 256, 0, stream>>>(src, result, feat, t);
    gemm(feat,  160,  4, b1,  bufA, 2048, 2048, 1);
    gemm(bufA, 2048,  5, b12, bufB, 2048, 2048, 1);
    gemm(bufB, 2048,  6, b2,  bufA, 1024, 1024, 1);
    gemm(bufA, 1024,  7, b22, bufB, 1024, 1024, 1);
    gemm(bufB, 1024,  8, b3,  bufA,  512,  512, 1);
    gemm(bufA,  512,  9, b4,  bufB,  512,  512, 0);
    gemm(bufB,  512, 10, b4d, result + (size_t)t * 69, 5865, 69, 0);
  }
}
